// CSNet_24721831756078
// MI455X (gfx1250) — matrix operand order repaired and checked by reading, not compiled, not run on hardware
//
#include <hip/hip_runtime.h>

typedef __attribute__((ext_vector_type(16))) _Float16 v16h;
typedef __attribute__((ext_vector_type(8)))  _Float16 v8h;
typedef __attribute__((ext_vector_type(8)))  float    v8f;
typedef __attribute__((ext_vector_type(4)))  unsigned int u32x4;

constexpr int IH = 256;
constexpr int IW = 256;
constexpr int NBATCH = 16;
constexpr int NPIX = NBATCH * IH * IW;      // 1,048,576 pixels
constexpr float BN_EPS = 1e-5f;

// ---------------------------------------------------------------------------
// utility: zero a float buffer (graph-capture-safe replacement for memset)
// ---------------------------------------------------------------------------
__global__ void zero_f32(float* p, int n) {
    int i = blockIdx.x * blockDim.x + threadIdx.x;
    if (i < n) p[i] = 0.f;
}

// ---------------------------------------------------------------------------
// Repack 10x [64,64,3,3] f32 OIHW weights into WMMA A-fragment-native f16:
//   wfrag[conv][kstep(9 taps x 2 cin-chunks)][cout_tile(4)][lane(32)][16 halves]
// Contiguous K order (matches B): lanes 0-15 hold M=lane, K = 0..15;
// lanes 16-31 hold M=lane-16, K = 16..31.
// ---------------------------------------------------------------------------
__global__ void repack_w(const float* __restrict__ c1w,
                         const float* __restrict__ c2w,
                         _Float16* __restrict__ wfrag) {
    int t = blockIdx.x * blockDim.x + threadIdx.x;
    if (t >= 10 * 18 * 4 * 32) return;
    int lane = t & 31;
    int ct   = (t >> 5) & 3;
    int ks   = (t >> 7) % 18;
    int cv   = t / (18 * 4 * 32);
    const float* w = ((cv & 1) ? c2w : c1w) + (size_t)(cv >> 1) * 64 * 64 * 9;
    int tap = ks >> 1, chunk = ks & 1;
    int m = lane & 15, khalf = lane >> 4;
    int cout = ct * 16 + m;
    _Float16* dst = wfrag + (size_t)cv * 18 * 4 * 32 * 16
                          + ((size_t)(ks * 4 + ct) * 32 + lane) * 16;
    for (int j = 0; j < 16; ++j) {
        int k = khalf * 16 + j;
        int cin = chunk * 32 + k;
        dst[j] = (_Float16)w[((size_t)cout * 64 + cin) * 9 + tap];
    }
}

// ---------------------------------------------------------------------------
// Sampling conv: x[16,1,256,256] (stride-32 32x32 kernel) -> cs[16,102,8,8]
// ---------------------------------------------------------------------------
__global__ void cs_kernel(const float* __restrict__ x,
                          const float* __restrict__ sw,
                          float* __restrict__ cs) {
    __shared__ float lx[1024];
    int bi = blockIdx.x;
    int j = bi & 7, i = (bi >> 3) & 7, b = bi >> 6;
    const float* xb = x + ((size_t)b * IH + i * 32) * IW + j * 32;
    for (int t = threadIdx.x; t < 1024; t += blockDim.x) {
        int p = t >> 5, q = t & 31;
        lx[t] = xb[(size_t)p * IW + q];
    }
    __syncthreads();
    int m = threadIdx.x;
    if (m < 102) {
        const float* wm = sw + (size_t)m * 1024;
        float acc = 0.f;
        for (int k = 0; k < 1024; ++k) acc = fmaf(lx[k], wm[k], acc);
        cs[(((size_t)b * 102 + m) * 8 + i) * 8 + j] = acc;
    }
}

// ---------------------------------------------------------------------------
// Init 1x1 conv (102->1024) fused with depth_to_space(32): writes y[16,256,256]
// ---------------------------------------------------------------------------
__global__ void init_kernel(const float* __restrict__ cs,
                            const float* __restrict__ iw,
                            const float* __restrict__ ib,
                            float* __restrict__ y) {
    __shared__ float lc[102];
    int bi = blockIdx.x;
    int j = bi & 7, i = (bi >> 3) & 7, b = bi >> 6;
    for (int t = threadIdx.x; t < 102; t += blockDim.x)
        lc[t] = cs[(((size_t)b * 102 + t) * 8 + i) * 8 + j];
    __syncthreads();
    for (int pq = threadIdx.x; pq < 1024; pq += blockDim.x) {
        const float* wrow = iw + (size_t)pq * 102;
        float acc = ib[pq];
        for (int m = 0; m < 102; ++m) acc = fmaf(lc[m], wrow[m], acc);
        int p = pq >> 5, q = pq & 31;
        y[((size_t)b * IH + (i * 32 + p)) * IW + (j * 32 + q)] = acc;
    }
}

// ---------------------------------------------------------------------------
// b1: 1->64 3x3 conv + ReLU, y (f32 NCHW, C=1) -> h (f16 NHWC)
// ---------------------------------------------------------------------------
__global__ void b1_kernel(const float* __restrict__ y,
                          const float* __restrict__ w,
                          const float* __restrict__ bias,
                          _Float16* __restrict__ h) {
    __shared__ float lw[64 * 9];
    __shared__ float lb[64];
    for (int t = threadIdx.x; t < 576; t += blockDim.x) lw[t] = w[t];
    for (int t = threadIdx.x; t < 64;  t += blockDim.x) lb[t] = bias[t];
    __syncthreads();
    int p = blockIdx.x * blockDim.x + threadIdx.x;
    if (p >= NPIX) return;
    int xw = p & (IW - 1);
    int yh = (p >> 8) & (IH - 1);
    int n  = p >> 16;
    float yv[9];
    for (int dy = 0; dy < 3; ++dy)
        for (int dx = 0; dx < 3; ++dx) {
            int yy = yh + dy - 1, xx = xw + dx - 1;
            yv[dy * 3 + dx] = (yy >= 0 && yy < IH && xx >= 0 && xx < IW)
                              ? y[((size_t)n * IH + yy) * IW + xx] : 0.f;
        }
    _Float16* hp = h + (size_t)p * 64;
    for (int c0 = 0; c0 < 64; c0 += 8) {
        v8h o;
        for (int c = 0; c < 8; ++c) {
            float acc = lb[c0 + c];
            const float* wr = lw + (c0 + c) * 9;
            for (int t = 0; t < 9; ++t) acc = fmaf(yv[t], wr[t], acc);
            o[c] = (_Float16)fmaxf(acc, 0.f);
        }
        *(v8h*)(hp + c0) = o;
    }
}

// ---------------------------------------------------------------------------
// Main 64->64 3x3 conv as implicit GEMM with v_wmma_f32_16x16x32_f16.
// in/out: NHWC f16. Workgroup: 256 threads / 8 waves computes a
// 32(x) x 8(y) x 64(cout) output block; each wave owns 2 y-rows x 16 x-cols
// and all 4 cout tiles => 8 accumulators, 144 WMMAs.
//
// Staging of the 10x34x64 f16 activation tile:
//   scsh == nullptr: global_load_async_to_lds_b128 (ASYNCcnt path), halo
//                    zero-filled with plain ds stores.
//   scsh != nullptr: fused epilogue-of-previous-layer: read raw conv output,
//                    apply BN scale/shift + PReLU in VGPRs, ds-store to LDS.
//                    (Saves the standalone 268MB bn_prelu pass per block.)
// BN sum/sumsq of THIS conv fused in the epilogue via wave32 shfl_xor
// butterfly + global f32 atomics.
// ---------------------------------------------------------------------------
__global__ void __launch_bounds__(256, 1)
conv3x3_wmma(const _Float16* __restrict__ in,
             const _Float16* __restrict__ wfrag,
             _Float16* __restrict__ out,
             float* __restrict__ stats,
             const float* __restrict__ scsh,
             const float* __restrict__ alpha) {
    __shared__ __align__(64) _Float16 la[10][36][64];  // [row y0-1..y0+8][x x0-1..x0+32][cin]
    __shared__ float lsc[128];
    int tid = threadIdx.x;

    int bid = blockIdx.x;
    int tx = bid & 7;             // 8 x-tiles of 32
    int ty = (bid >> 3) & 31;     // 32 y-tiles of 8
    int n  = bid >> 8;            // batch
    int x0 = tx * 32, y0 = ty * 8;

    if (scsh) {
        // ---- fused BN+PReLU staging (input is a raw conv output) ----
        if (tid < 128) lsc[tid] = scsh[tid];
        __syncthreads();
        float a = alpha[0];
        for (int task = tid; task < 340 * 8; task += 256) {
            int part = task & 7;
            int s = task >> 3;
            int r = s / 34, c = s % 34;
            int yg = y0 + r - 1, xg = x0 + c - 1;
            int c0 = part * 8;
            v8h o = (v8h){(_Float16)0, (_Float16)0, (_Float16)0, (_Float16)0,
                          (_Float16)0, (_Float16)0, (_Float16)0, (_Float16)0};
            if (yg >= 0 && yg < IH && xg >= 0 && xg < IW) {
                v8h xv = *(const v8h*)(in + (((size_t)n * IH + yg) * IW + xg) * 64 + c0);
#pragma unroll
                for (int j = 0; j < 8; ++j) {
                    float v = (float)xv[j] * lsc[c0 + j] + lsc[64 + c0 + j];
                    o[j] = (_Float16)(v > 0.f ? v : a * v);
                }
            }
            *(v8h*)(&la[r][c][c0]) = o;
        }
    } else {
        // ---- async direct-to-LDS staging (input is a finished activation) ----
        for (int task = tid; task < 340 * 8; task += 256) {
            int part = task & 7;
            int s = task >> 3;
            int r = s / 34, c = s % 34;
            int yg = y0 + r - 1, xg = x0 + c - 1;
            if (yg >= 0 && yg < IH && xg >= 0 && xg < IW) {
                unsigned ldsa = (unsigned)(size_t)&la[r][c][part * 8];
                unsigned long long ga = (unsigned long long)(size_t)
                    (in + (((size_t)n * IH + yg) * IW + xg) * 64 + part * 8);
                asm volatile("global_load_async_to_lds_b128 %0, %1, off"
                             :: "v"(ldsa), "v"(ga) : "memory");
            } else {
                *(u32x4*)(&la[r][c][part * 8]) = (u32x4){0u, 0u, 0u, 0u};
            }
        }
        asm volatile("s_wait_asynccnt 0" ::: "memory");
    }
    __syncthreads();

    int wave = tid >> 5, lane = tid & 31;
    int xh = wave & 1;            // 16-pixel x-half
    int yp = wave >> 1;           // y-pair 0..3
    int col = lane & 15, kh = lane >> 4;
    int xc = xh * 16 + col;       // x within tile (before dx shift)

    const v16h* wp = (const v16h*)wfrag;

    v8f acc[4][2];
#pragma unroll
    for (int ct = 0; ct < 4; ++ct)
#pragma unroll
        for (int t = 0; t < 2; ++t)
            acc[ct][t] = (v8f){0.f, 0.f, 0.f, 0.f, 0.f, 0.f, 0.f, 0.f};

#pragma unroll
    for (int ks = 0; ks < 18; ++ks) {
        const int tap = ks >> 1;
        const int dy = tap / 3, dx = tap % 3;
        const int ch = (ks & 1) * 32;
        // B-fragments for the two y-rows: lane holds K = kh*16..+15 (contig cin)
        v16h bfr[2];
#pragma unroll
        for (int t = 0; t < 2; ++t)
            bfr[t] = *(const v16h*)(&la[yp * 2 + t + dy][xc + dx][ch + kh * 16]);
#pragma unroll
        for (int ct = 0; ct < 4; ++ct) {
            v16h a = wp[(ks * 4 + ct) * 32 + lane];
#pragma unroll
            for (int t = 0; t < 2; ++t)
                acc[ct][t] = __builtin_amdgcn_wmma_f32_16x16x32_f16(
                    false, a, false, bfr[t], (short)0, acc[ct][t], false, false);
        }
    }

    // Epilogue: lane holds M = kh*8 + i, N = col for each tile.
    int x = x0 + xc;
#pragma unroll
    for (int ct = 0; ct < 4; ++ct) {
        int mbase = ct * 16 + kh * 8;
#pragma unroll
        for (int t = 0; t < 2; ++t) {
            int yy = y0 + yp * 2 + t;
            _Float16* op = out + (((size_t)n * IH + yy) * IW + x) * 64 + mbase;
            v8h ov;
#pragma unroll
            for (int i = 0; i < 8; ++i) ov[i] = (_Float16)acc[ct][t][i];
            *(v8h*)op = ov;
        }
    }

    // BN stats: pre-sum the two y-tiles, butterfly-reduce over the 16 N-lanes
    // (wave32 shfl), then one global f32 atomic per channel from col==0 lanes.
#pragma unroll
    for (int ct = 0; ct < 4; ++ct)
#pragma unroll
        for (int i = 0; i < 8; ++i) {
            float s = acc[ct][0][i] + acc[ct][1][i];
            float q = fmaf(acc[ct][0][i], acc[ct][0][i],
                           acc[ct][1][i] * acc[ct][1][i]);
#pragma unroll
            for (int m = 8; m >= 1; m >>= 1) {
                s += __shfl_xor(s, m, 32);
                q += __shfl_xor(q, m, 32);
            }
            if (col == 0) {
                int cch = ct * 16 + kh * 8 + i;
                atomicAdd(&stats[cch], s);
                atomicAdd(&stats[64 + cch], q);
            }
        }
}

// ---------------------------------------------------------------------------
// BN finalize: stats -> per-channel scale/shift (training-mode, biased var)
// ---------------------------------------------------------------------------
__global__ void bn_finalize(const float* __restrict__ stats,
                            const float* __restrict__ g,
                            const float* __restrict__ b,
                            float* __restrict__ scsh) {
    int c = threadIdx.x;
    if (c >= 64) return;
    const float inv = 1.f / (float)NPIX;
    float mean = stats[c] * inv;
    float var  = stats[64 + c] * inv - mean * mean;
    float r    = rsqrtf(var + BN_EPS);
    float sc   = g[c] * r;
    scsh[c]      = sc;
    scsh[64 + c] = b[c] - mean * sc;
}

// ---------------------------------------------------------------------------
// BN + residual add (in-place into skip): skip = skip + bn(conv)
// ---------------------------------------------------------------------------
__global__ void bn_add(const _Float16* __restrict__ conv,
                       const float* __restrict__ scsh,
                       _Float16* __restrict__ skip) {
    __shared__ float ls[128];
    if (threadIdx.x < 128) ls[threadIdx.x] = scsh[threadIdx.x];
    __syncthreads();
    size_t i = ((size_t)blockIdx.x * blockDim.x + threadIdx.x) * 8;
    int c0 = (int)(i & 63);
    v8h xv = *(const v8h*)(conv + i);
    v8h sv = *(const v8h*)(skip + i);
    v8h o;
#pragma unroll
    for (int j = 0; j < 8; ++j) {
        float v = (float)xv[j] * ls[c0 + j] + ls[64 + c0 + j] + (float)sv[j];
        o[j] = (_Float16)v;
    }
    *(v8h*)(skip + i) = o;
}

// ---------------------------------------------------------------------------
// Final 64->1 3x3 conv + bias, plus skip with y: d_out = y + conv(h)
// ---------------------------------------------------------------------------
__global__ void out_conv(const _Float16* __restrict__ h,
                         const float* __restrict__ w,
                         const float* __restrict__ bias,
                         const float* __restrict__ y,
                         float* __restrict__ outp) {
    __shared__ float lw[576];
    for (int t = threadIdx.x; t < 576; t += blockDim.x) lw[t] = w[t];
    __syncthreads();
    int p = blockIdx.x * blockDim.x + threadIdx.x;
    if (p >= NPIX) return;
    int xw = p & (IW - 1), yh = (p >> 8) & (IH - 1), n = p >> 16;
    float acc = bias[0];
    for (int dy = 0; dy < 3; ++dy) {
        int yy = yh + dy - 1;
        if (yy < 0 || yy >= IH) continue;
        for (int dx = 0; dx < 3; ++dx) {
            int xx = xw + dx - 1;
            if (xx < 0 || xx >= IW) continue;
            const _Float16* hp = h + (((size_t)n * IH + yy) * IW + xx) * 64;
            int t = dy * 3 + dx;
            for (int c = 0; c < 64; ++c)
                acc = fmaf((float)hp[c], lw[c * 9 + t], acc);
        }
    }
    outp[p] = y[p] + acc;
}

// ---------------------------------------------------------------------------
extern "C" void kernel_launch(void* const* d_in, const int* in_sizes, int n_in,
                              void* d_out, int out_size, void* d_ws, size_t ws_size,
                              hipStream_t stream) {
    (void)in_sizes; (void)n_in; (void)out_size; (void)ws_size;

    const float* x      = (const float*)d_in[0];
    const float* samp_w = (const float*)d_in[1];
    const float* init_w = (const float*)d_in[2];
    const float* init_b = (const float*)d_in[3];
    const float* b1_w   = (const float*)d_in[4];
    const float* b1_b   = (const float*)d_in[5];
    const float* c1w    = (const float*)d_in[6];
    // d_in[7] = rb_c1_b: pre-BN bias cancels under training-mode BN (and is 0)
    const float* g1     = (const float*)d_in[8];
    const float* be1    = (const float*)d_in[9];
    const float* prelu  = (const float*)d_in[10];
    const float* c2w    = (const float*)d_in[11];
    // d_in[12] = rb_c2_b: same
    const float* g2     = (const float*)d_in[13];
    const float* be2    = (const float*)d_in[14];
    const float* outw   = (const float*)d_in[15];
    const float* outb   = (const float*)d_in[16];

    char* ws = (char*)d_ws;
    size_t off = 0;
    auto alloc = [&](size_t bytes) -> void* {
        void* p = ws + off;
        off = (off + bytes + 255) & ~(size_t)255;
        return p;
    };
    _Float16* actA  = (_Float16*)alloc((size_t)NPIX * 64 * 2);   // h (skip)
    _Float16* actB  = (_Float16*)alloc((size_t)NPIX * 64 * 2);   // conv2 raw out
    _Float16* convC = (_Float16*)alloc((size_t)NPIX * 64 * 2);   // conv1 raw out
    float*    yb    = (float*)alloc((size_t)NPIX * 4);           // y (f32)
    float*    cs    = (float*)alloc((size_t)16 * 102 * 64 * 4);
    _Float16* wfrag = (_Float16*)alloc((size_t)10 * 18 * 4 * 32 * 16 * 2);
    float*    stats = (float*)alloc(10 * 128 * 4);
    float*    scsh  = (float*)alloc(10 * 128 * 4);

    zero_f32<<<(1280 + 255) / 256, 256, 0, stream>>>(stats, 1280);
    repack_w<<<(10 * 18 * 4 * 32 + 255) / 256, 256, 0, stream>>>(c1w, c2w, wfrag);
    cs_kernel<<<1024, 256, 0, stream>>>(x, samp_w, cs);
    init_kernel<<<1024, 256, 0, stream>>>(cs, init_w, init_b, yb);
    b1_kernel<<<NPIX / 256, 256, 0, stream>>>(yb, b1_w, b1_b, actA);

    const size_t WFC = (size_t)18 * 4 * 32 * 16;
    for (int i = 0; i < 5; ++i) {
        // conv1: actA -> convC (async staging), stats[2i]
        conv3x3_wmma<<<4096, 256, 0, stream>>>(
            actA, wfrag + (size_t)(2 * i) * WFC, convC, stats + (2 * i) * 128,
            nullptr, nullptr);
        bn_finalize<<<1, 64, 0, stream>>>(
            stats + (2 * i) * 128, g1 + i * 64, be1 + i * 64, scsh + (2 * i) * 128);

        // conv2: convC (BN1+PReLU fused into staging) -> actB, stats[2i+1]
        conv3x3_wmma<<<4096, 256, 0, stream>>>(
            convC, wfrag + (size_t)(2 * i + 1) * WFC, actB, stats + (2 * i + 1) * 128,
            scsh + (2 * i) * 128, prelu + i);
        bn_finalize<<<1, 64, 0, stream>>>(
            stats + (2 * i + 1) * 128, g2 + i * 64, be2 + i * 64, scsh + (2 * i + 1) * 128);

        // residual: actA += bn2(actB)
        bn_add<<<32768, 256, 0, stream>>>(
            actB, scsh + (2 * i + 1) * 128, actA);
    }
    out_conv<<<NPIX / 256, 256, 0, stream>>>(actA, outw, outb, yb, (float*)d_out);
}
